// DynamicCostVolumeRefinement_84061099917709
// MI455X (gfx1250) — matrix operand order repaired and checked by reading, not compiled, not run on hardware
//
#include <hip/hip_runtime.h>
#include <hip/hip_bf16.h>
#include <stdint.h>
#include <stddef.h>

// ---------------------------------------------------------------------------
// Types for CDNA5 WMMA (wave32, 16x16x32 bf16 -> f32)
// ---------------------------------------------------------------------------
typedef __attribute__((ext_vector_type(16))) __bf16 v16bf;
typedef __attribute__((ext_vector_type(8)))  float  v8f;

// TDM descriptor groups
typedef __attribute__((ext_vector_type(4))) unsigned tdm_u32x4;
typedef __attribute__((ext_vector_type(8))) int      tdm_i32x8;
typedef __attribute__((ext_vector_type(4))) int      tdm_i32x4;

static __device__ __forceinline__ unsigned short f2bf(float f) {
  unsigned u = __float_as_uint(f);
  unsigned r = u + 0x7FFFu + ((u >> 16) & 1u);   // round-to-nearest-even
  return (unsigned short)(r >> 16);
}

// ---------------------------------------------------------------------------
// TDM: 2D tile load Global -> LDS (bf16 elements), row-padded in LDS.
//   tile: tile1 rows x tile0 elems, global row stride = stride0 elems.
//   tensor dims (td0 cols, td1 rows) enable hardware OOB zero-fill for tails.
//   LDS pad: pad_interval=4 (32 DWORDs = one 64-elem row), pad_amount=0
//   (1 DWORD = 2 ushorts) -> LDS row stride 66 ushorts (= BS_STRIDE).
// ---------------------------------------------------------------------------
static __device__ __forceinline__
void tdm_load_2d_bf16(unsigned lds_addr, const void* gaddr,
                      int td0, int td1, int tile0, int tile1, int stride0) {
  unsigned long long ga = (unsigned long long)(size_t)gaddr;
  tdm_u32x4 g0;
  g0[0] = 1u;                                            // count=1 (valid D#)
  g0[1] = lds_addr;                                      // LDS byte address
  g0[2] = (unsigned)(ga & 0xFFFFFFFFull);                // global_addr[31:0]
  g0[3] = (unsigned)((ga >> 32) & 0x1FFFFFFull)          // global_addr[56:32]
        | (2u << 30);                                    // type=2 ("image")
  tdm_i32x8 g1;
  g1[0] = (1 << 16)                                      // data_size = 2 bytes
        | (1 << 20)                                      // pad_enable
        | (4 << 22);                                     // pad_interval = 32 DW
                                                         // pad_amount = 0 (1 DW)
  g1[1] = (int)(((unsigned)td0 & 0xFFFFu) << 16);        // tensor_dim0[15:0]
  g1[2] = (int)((((unsigned)td0 >> 16) & 0xFFFFu)        // tensor_dim0[31:16]
        | (((unsigned)td1 & 0xFFFFu) << 16));            // tensor_dim1[15:0]
  g1[3] = (int)((((unsigned)td1 >> 16) & 0xFFFFu)        // tensor_dim1[31:16]
        | (((unsigned)tile0 & 0xFFFFu) << 16));          // tile_dim0
  g1[4] = (int)((unsigned)tile1 & 0xFFFFu);              // tile_dim1 (tile_dim2=0)
  g1[5] = stride0;                                       // tensor_dim0_stride[31:0]
  g1[6] = 0;
  g1[7] = 0;
  tdm_i32x4 gz = {0, 0, 0, 0};
#if defined(__clang_major__) && (__clang_major__ >= 23)
  tdm_i32x8 gz8 = {0, 0, 0, 0, 0, 0, 0, 0};
  __builtin_amdgcn_tensor_load_to_lds(g0, g1, gz, gz, gz8, 0);
#else
  __builtin_amdgcn_tensor_load_to_lds(g0, g1, gz, gz, 0);
#endif
}

// ---------------------------------------------------------------------------
// Kernel 1: build U-Net input = concat(left_clues - bilinear_warp(right_clues),
//           disp) as bf16 [B,132,H,W]
// ---------------------------------------------------------------------------
__global__ __launch_bounds__(256)
void build_unet_input(const float* __restrict__ lf, const float* __restrict__ li,
                      const float* __restrict__ rf, const float* __restrict__ ri,
                      const float* __restrict__ disp,
                      unsigned short* __restrict__ xin,
                      int Bn, int H, int W) {
  int idx = blockIdx.x * blockDim.x + threadIdx.x;
  int HW = H * W;
  if (idx >= Bn * HW) return;
  int b = idx / HW, s = idx - b * HW;
  int h = s / W, w = s - h * W;
  float d = disp[(size_t)b * HW + s];
  float xs = ((float)w - d) * ((float)W / (W - 1.f)) - 0.5f;
  xs = fminf(fmaxf(xs, 0.f), (float)(W - 1));
  float ys = (float)h * ((float)H / (H - 1.f)) - 0.5f;
  ys = fminf(fmaxf(ys, 0.f), (float)(H - 1));
  float x0f = floorf(xs), wx = xs - x0f;
  float y0f = floorf(ys), wy = ys - y0f;
  int ix0 = (int)x0f, ix1 = min(ix0 + 1, W - 1);
  int iy0 = (int)y0f, iy1 = min(iy0 + 1, H - 1);
  for (int c = 0; c < 131; ++c) {
    const float* src = (c < 128) ? (rf + ((size_t)(b * 128 + c)) * HW)
                                 : (ri + ((size_t)(b * 3 + (c - 128))) * HW);
    float tl = src[iy0 * W + ix0], tr = src[iy0 * W + ix1];
    float bl = src[iy1 * W + ix0], br = src[iy1 * W + ix1];
    float vt = (1.f - wx) * tl + wx * tr;
    float vb = (1.f - wx) * bl + wx * br;
    float wp = (1.f - wy) * vt + wy * vb;
    float lc = (c < 128) ? lf[((size_t)(b * 128 + c)) * HW + s]
                         : li[((size_t)(b * 3 + (c - 128))) * HW + s];
    xin[((size_t)(b * 132 + c)) * HW + s] = f2bf(lc - wp);
  }
  xin[((size_t)(b * 132 + 131)) * HW + s] = f2bf(d);
}

// ---------------------------------------------------------------------------
// Kernel 2: weight conversion f32 -> bf16 [K = Cin*9][Cout]
// ---------------------------------------------------------------------------
__global__ __launch_bounds__(256)
void wcvt_kernel(const float* __restrict__ w, unsigned short* __restrict__ dst,
                 int Cin, int Cout, int transposed) {
  int idx = blockIdx.x * blockDim.x + threadIdx.x;
  int K = Cin * 9;
  if (idx >= K * Cout) return;
  int k = idx / Cout, co = idx - k * Cout;
  int ci = k / 9, t = k - ci * 9;
  int kh = t / 3, kw = t - kh * 3;
  float v;
  if (!transposed) v = w[(((size_t)co * Cin + ci) * 3 + kh) * 3 + kw];
  else             v = w[(((size_t)ci * Cout + co) * 3 + (2 - kh)) * 3 + (2 - kw)];
  dst[idx] = f2bf(v);
}

// ---------------------------------------------------------------------------
// Kernel 3: implicit-GEMM conv via WMMA bf16 + TDM weight staging.
//   GEMM:  [Mtot = B*Hout*Wout] x [N = Cout]  with K = Cin*9
//   Block: 128 threads = 4 wave32s. Block tile 64(M) x 64(N):
//     wave wv owns M-subtile rows [m0 + 16*wv, +16); computes FOUR 16x16
//     accumulators across the 64-wide N tile -> 4 v_wmma per K-step, one
//     shared A fragment per wave.
//   B tile (weights) is DMA'd by the Tensor Data Mover (wave 0 issues one
//   descriptor; TDM OOB zero-fill covers K/N tails; LDS pad gives stride 66),
//   overlapping with the VALU im2col A-gather.
//   transposed==1 implements lhs-dilated conv (ConvTranspose2d k3 s2 p1 op1).
// ---------------------------------------------------------------------------
#define AS_STRIDE 34
#define BS_STRIDE 66

__global__ __launch_bounds__(128)
void conv_wmma_kernel(const unsigned short* __restrict__ in,   // bf16 [B,Cin,Hin,Win]
                      const unsigned short* __restrict__ wb,   // bf16 [K][Cout]
                      float* __restrict__ out,                 // f32 [B,Cout,Hout,Wout]
                      int Bn, int Cin, int Hin, int Win,
                      int Cout, int Hout, int Wout,
                      int stride, int transposed) {
  __shared__ unsigned short As[64 * AS_STRIDE];
  __shared__ unsigned short Bs[32 * BS_STRIDE];
  const int K = Cin * 9;
  const int HWo = Hout * Wout;
  const int Mtot = Bn * HWo;
  const int m0 = blockIdx.x * 64;
  const int n0 = blockIdx.y * 64;
  const int tid = threadIdx.x;
  const int lane = tid & 31;           // wave32
  const int wv = tid >> 5;             // wave id 0..3
  const unsigned bs_lds = (unsigned)(size_t)&Bs[0];   // LDS aperture: addr[31:0]

  union { v8f v; float f[8]; } acc[4];
#pragma unroll
  for (int j = 0; j < 4; ++j)
#pragma unroll
    for (int r = 0; r < 8; ++r) acc[j].f[r] = 0.f;

  for (int k0 = 0; k0 < K; k0 += 32) {
    // ---- issue TDM for B tile (weights): 32 rows x 64 cols, wave 0 only ----
    if (tid < 32) {
      tdm_load_2d_bf16(bs_lds, wb + (size_t)k0 * Cout + n0,
                       /*td0=*/Cout - n0, /*td1=*/K - k0,
                       /*tile0=*/64, /*tile1=*/32, /*stride0=*/Cout);
    }
    // ---- stage A tile (64 x 32 bf16, im2col gather; overlaps with TDM) ----
    for (int e = tid; e < 64 * 32; e += 128) {
      int mi = e >> 5, kj = e & 31;
      int m = m0 + mi, k = k0 + kj;
      unsigned short v = 0;
      if (m < Mtot && k < K) {
        int b = m / HWo; int rem = m - b * HWo;
        int oh = rem / Wout; int ow = rem - oh * Wout;
        int ci = k / 9; int t = k - ci * 9;
        int kh = t / 3, kw = t - kh * 3;
        if (!transposed) {
          int iy = oh * stride + kh - 1;
          int ix = ow * stride + kw - 1;
          if (iy >= 0 && iy < Hin && ix >= 0 && ix < Win)
            v = in[(((size_t)b * Cin + ci) * Hin + iy) * Win + ix];
        } else {
          int iyd = oh + kh - 1, ixd = ow + kw - 1;
          if (iyd >= 0 && ixd >= 0 && !(iyd & 1) && !(ixd & 1)) {
            int iy = iyd >> 1, ix = ixd >> 1;
            if (iy < Hin && ix < Win)
              v = in[(((size_t)b * Cin + ci) * Hin + iy) * Win + ix];
          }
        }
      }
      As[mi * AS_STRIDE + kj] = v;
    }
    if (tid < 32) __builtin_amdgcn_s_wait_tensorcnt(0);  // B tile landed in LDS
    __syncthreads();

    // ---- pack A fragment (CDNA5 16-bit A layout), shared across 4 WMMAs ----
    union { v16bf v; unsigned u[8]; } af;
    {
      int m = (wv << 4) + (lane & 15);
      int hh = (lane >= 16) ? 8 : 0;
#pragma unroll
      for (int vv = 0; vv < 8; ++vv) {
        int kk = 2 * vv + ((vv >= 4) ? 8 : 0) + hh;
        unsigned lo = As[m * AS_STRIDE + kk];
        unsigned hi = As[m * AS_STRIDE + kk + 1];
        af.u[vv] = lo | (hi << 16);
      }
    }
    // ---- 4 N-subtiles: pack B frag + WMMA each ----
    int nlane = lane & 15;
    int kb = (lane >= 16) ? 8 : 0;
#pragma unroll
    for (int j = 0; j < 4; ++j) {
      union { v16bf v; unsigned u[8]; } bfm;
      int n = j * 16 + nlane;
#pragma unroll
      for (int vv = 0; vv < 8; ++vv) {
        unsigned lo = Bs[(kb + 2 * vv + ((vv >= 4) ? 8 : 0)) * BS_STRIDE + n];
        unsigned hi = Bs[(kb + 2 * vv + ((vv >= 4) ? 8 : 0) + 1) * BS_STRIDE + n];
        bfm.u[vv] = lo | (hi << 16);
      }
      acc[j].v = __builtin_amdgcn_wmma_f32_16x16x32_bf16(
          false, af.v, false, bfm.v, (short)0, acc[j].v, false, false);
    }
    __syncthreads();   // protect Bs/As before next K-step's TDM/gather
  }

  // ---- epilogue: 16x16 f32 C/D layout -> NCHW f32, 4 subtiles ----
  {
    int nlane = lane & 15;
    int mh = (lane >= 16) ? 8 : 0;
#pragma unroll
    for (int j = 0; j < 4; ++j) {
      int cout = n0 + j * 16 + nlane;
      if (cout < Cout) {
#pragma unroll
        for (int r = 0; r < 8; ++r) {
          int m = m0 + (wv << 4) + r + mh;
          if (m < Mtot) {
            int b = m / HWo; int rem = m - b * HWo;
            out[((size_t)b * Cout + cout) * HWo + rem] = acc[j].f[r];
          }
        }
      }
    }
  }
}

// ---------------------------------------------------------------------------
// Kernel 4: BatchNorm batch statistics (biased var, matches jnp.var)
// ---------------------------------------------------------------------------
__global__ __launch_bounds__(256)
void bn_stats(const float* __restrict__ x, float* __restrict__ mean,
              float* __restrict__ var, int Bn, int C, int HW) {
  int c = blockIdx.x;
  __shared__ float s1[256], s2[256];
  float a = 0.f, q = 0.f;
  int N = Bn * HW;
  for (int i = threadIdx.x; i < N; i += 256) {
    int b = i / HW, s = i - b * HW;
    float v = x[((size_t)b * C + c) * HW + s];
    a += v; q += v * v;
  }
  s1[threadIdx.x] = a; s2[threadIdx.x] = q;
  __syncthreads();
  for (int off = 128; off > 0; off >>= 1) {
    if (threadIdx.x < off) {
      s1[threadIdx.x] += s1[threadIdx.x + off];
      s2[threadIdx.x] += s2[threadIdx.x + off];
    }
    __syncthreads();
  }
  if (threadIdx.x == 0) {
    float m = s1[0] / (float)N;
    mean[c] = m;
    var[c] = s2[0] / (float)N - m * m;
  }
}

// ---------------------------------------------------------------------------
// Kernel 5: BN normalize + ReLU/LeakyReLU, emit bf16 (optionally to 2 places
// for the skip-connection concat buffer)
// ---------------------------------------------------------------------------
__global__ __launch_bounds__(256)
void bn_norm_act(const float* __restrict__ x, const float* __restrict__ mean,
                 const float* __restrict__ var, const float* __restrict__ g,
                 const float* __restrict__ bt,
                 unsigned short* __restrict__ dst, int dstC, int c_off,
                 unsigned short* __restrict__ dst2, int dst2C, int c_off2,
                 int Bn, int C, int HW, int leaky) {
  int idx = blockIdx.x * blockDim.x + threadIdx.x;
  if (idx >= Bn * C * HW) return;
  int s = idx % HW;
  int c = (idx / HW) % C;
  int b = idx / (HW * C);
  float v = x[idx];
  float y = (v - mean[c]) * rsqrtf(var[c] + 1e-5f) * g[c] + bt[c];
  y = (y >= 0.f) ? y : (leaky ? 0.2f * y : 0.f);
  unsigned short bv = f2bf(y);
  dst[((size_t)b * dstC + c_off + c) * HW + s] = bv;
  if (dst2) dst2[((size_t)b * dst2C + c_off2 + c) * HW + s] = bv;
}

// ---------------------------------------------------------------------------
// Kernel 6: offset head epilogue: +bias, ReLU -> lower / upper (f32, to d_out)
// ---------------------------------------------------------------------------
__global__ __launch_bounds__(256)
void off_bias_relu(const float* __restrict__ raw, const float* __restrict__ bias,
                   float* __restrict__ lower, float* __restrict__ upper,
                   int Bn, int HW) {
  int idx = blockIdx.x * blockDim.x + threadIdx.x;
  if (idx >= Bn * HW) return;
  int b = idx / HW, s = idx - b * HW;
  float l = raw[((size_t)(b * 2 + 0)) * HW + s] + bias[0];
  float u = raw[((size_t)(b * 2 + 1)) * HW + s] + bias[1];
  lower[idx] = fmaxf(l, 0.f);
  upper[idx] = fmaxf(u, 0.f);
}

// ---------------------------------------------------------------------------
// Kernel 7: global max reductions (non-negative floats -> atomicMax on bits)
// ---------------------------------------------------------------------------
__global__ void zero_scal(float* p) { if (threadIdx.x < 2) p[threadIdx.x] = 0.f; }

__global__ __launch_bounds__(256)
void reduce_max(const float* __restrict__ x, int n, float* __restrict__ out) {
  __shared__ float sm[256];
  float m = 0.f;
  for (int i = blockIdx.x * 256 + threadIdx.x; i < n; i += gridDim.x * 256)
    m = fmaxf(m, x[i]);
  sm[threadIdx.x] = m;
  __syncthreads();
  for (int off = 128; off > 0; off >>= 1) {
    if (threadIdx.x < off) sm[threadIdx.x] = fmaxf(sm[threadIdx.x], sm[threadIdx.x + off]);
    __syncthreads();
  }
  if (threadIdx.x == 0) atomicMax((unsigned*)out, __float_as_uint(sm[0]));
}

// ---------------------------------------------------------------------------
// Kernel 8: cost volume — per pixel, loop R candidates; 128-ch dot product of
// left feature with nearest-warped right feature; write acv + mask.
// ---------------------------------------------------------------------------
__global__ __launch_bounds__(256)
void cost_volume(const float* __restrict__ lf, const float* __restrict__ rf,
                 const float* __restrict__ disp, const float* __restrict__ lower,
                 const float* __restrict__ upper, const float* __restrict__ scal,
                 float* __restrict__ acv, float* __restrict__ mask,
                 int Bn, int C, int H, int W, int R) {
  int idx = blockIdx.x * blockDim.x + threadIdx.x;
  int HW = H * W;
  if (idx >= Bn * HW) return;
  int b = idx / HW, s = idx - b * HW;
  int h = s / W, w = s - h * W;
  float max_lb = rintf(scal[0]);
  float rd = rintf(disp[idx]);
  float vlo = fminf(fmaxf(rintf(lower[idx]), 0.f), (float)(W - 1));
  float vhi = fminf(fmaxf(rintf(upper[idx]), 0.f), (float)(W - 1));
  float ysc = fminf(fmaxf((float)h * ((float)H / (H - 1.f)) - 0.5f, 0.f), (float)(H - 1));
  int iy = (int)rintf(ysc);
  const float* lrow = lf + (size_t)b * C * HW + s;
  const float* rbase = rf + (size_t)b * C * HW + (size_t)iy * W;
  for (int i = 0; i < R; ++i) {
    float cand = rd + (float)i - max_lb;
    float candc = fmaxf(cand, 0.f);
    bool gate = (cand >= 0.f) && (((float)w - candc) >= 0.f);
    float vol = 0.f;
    if (gate) {
      float xs = ((float)w - candc) * ((float)W / (W - 1.f)) - 0.5f;
      xs = fminf(fmaxf(xs, 0.f), (float)(W - 1));
      int ix = (int)rintf(xs);
      const float* rrow = rbase + ix;
      __builtin_prefetch(rrow, 0, 1);          // global_prefetch_b8
      float a0 = 0.f, a1 = 0.f, a2 = 0.f, a3 = 0.f;
      for (int c = 0; c < C; c += 4) {
        a0 += lrow[(size_t)c * HW]       * rrow[(size_t)c * HW];
        a1 += lrow[(size_t)(c + 1) * HW] * rrow[(size_t)(c + 1) * HW];
        a2 += lrow[(size_t)(c + 2) * HW] * rrow[(size_t)(c + 2) * HW];
        a3 += lrow[(size_t)(c + 3) * HW] * rrow[(size_t)(c + 3) * HW];
      }
      vol = (a0 + a1 + a2 + a3) / (float)C;
    }
    float mf = ((cand >= rd - vlo) && (cand <= rd + vhi)) ? 1.f : 0.f;
    size_t o = ((size_t)(b * R + i)) * HW + s;
    acv[o] = vol * mf - 1000.f * (1.f - mf);
    mask[o] = mf;
  }
}

// ---------------------------------------------------------------------------
// Kernel 9: softmax over R + expected residual
// ---------------------------------------------------------------------------
__global__ __launch_bounds__(256)
void softmax_residual(const float* __restrict__ acv, const float* __restrict__ scal,
                      float* __restrict__ residual, int Bn, int HW, int R) {
  int idx = blockIdx.x * blockDim.x + threadIdx.x;
  if (idx >= Bn * HW) return;
  int b = idx / HW, s = idx - b * HW;
  float max_lb = rintf(scal[0]);
  const float* a = acv + (size_t)b * R * HW + s;
  float mx = -1e30f;
  for (int i = 0; i < R; ++i) mx = fmaxf(mx, a[(size_t)i * HW]);
  float sum = 0.f, wsum = 0.f;
  for (int i = 0; i < R; ++i) {
    float e = expf(a[(size_t)i * HW] - mx);
    sum += e;
    wsum += e * ((float)i - max_lb);
  }
  residual[idx] = wsum / sum;
}

// ---------------------------------------------------------------------------
// Orchestration
// ---------------------------------------------------------------------------
extern "C" void kernel_launch(void* const* d_in, const int* in_sizes, int n_in,
                              void* d_out, int out_size, void* d_ws, size_t ws_size,
                              hipStream_t stream) {
  const int B = 4, C = 128, H = 96, W = 320;
  const int HW = H * W, BHW = B * HW;
  const int H1 = 48, W1 = 160, HW1 = H1 * W1;   // stride-2 level 1
  const int H2 = 24, W2 = 80,  HW2 = H2 * W2;   // stride-2 level 2

  const float* lf   = (const float*)d_in[0];
  const float* rf   = (const float*)d_in[1];
  const float* disp = (const float*)d_in[2];
  const float* li   = (const float*)d_in[3];
  const float* ri   = (const float*)d_in[4];
  // params in dict insertion order
  const float* c1a_w = (const float*)d_in[5];  const float* c1a_g = (const float*)d_in[6];  const float* c1a_b = (const float*)d_in[7];
  const float* c1b_w = (const float*)d_in[8];  const float* c1b_g = (const float*)d_in[9];  const float* c1b_b = (const float*)d_in[10];
  const float* c2a_w = (const float*)d_in[11]; const float* c2a_g = (const float*)d_in[12]; const float* c2a_b = (const float*)d_in[13];
  const float* c2b_w = (const float*)d_in[14]; const float* c2b_g = (const float*)d_in[15]; const float* c2b_b = (const float*)d_in[16];
  const float* up1_w = (const float*)d_in[17]; const float* up1_g = (const float*)d_in[18]; const float* up1_b = (const float*)d_in[19];
  const float* up2_w = (const float*)d_in[20]; const float* up2_g = (const float*)d_in[21]; const float* up2_b = (const float*)d_in[22];
  const float* off_w = (const float*)d_in[23]; const float* off_b = (const float*)d_in[24];

  int R = (out_size - 3 * BHW) / (2 * BHW);
  if (R < 1) R = 1;

  float* out_res  = (float*)d_out;
  float* out_acv  = out_res + BHW;
  float* out_mask = out_acv + (size_t)R * BHW;
  float* out_low  = out_mask + (size_t)R * BHW;
  float* out_up   = out_low + BHW;

  // ---- workspace carve ----
  char* p = (char*)d_ws;
  auto carve = [&](size_t bytes) -> char* {
    char* r = p; p += (bytes + 255) & ~(size_t)255; return r;
  };
  unsigned short* xin    = (unsigned short*)carve((size_t)B * 132 * HW * 2);
  unsigned short* wb_c1a = (unsigned short*)carve((size_t)1188 * 64 * 2);
  unsigned short* wb_c1b = (unsigned short*)carve((size_t)576 * 64 * 2);
  unsigned short* wb_c2a = (unsigned short*)carve((size_t)576 * 128 * 2);
  unsigned short* wb_c2b = (unsigned short*)carve((size_t)1152 * 128 * 2);
  unsigned short* wb_up1 = (unsigned short*)carve((size_t)1152 * 64 * 2);
  unsigned short* wb_up2 = (unsigned short*)carve((size_t)1152 * 64 * 2);
  unsigned short* wb_off = (unsigned short*)carve((size_t)576 * 2 * 2);
  float*          raw    = (float*)carve((size_t)B * 64 * HW * 4);   // largest raw output
  unsigned short* a_f1a  = (unsigned short*)carve((size_t)B * 64 * HW1 * 2);
  unsigned short* a_f1b  = (unsigned short*)carve((size_t)B * 64 * HW1 * 2);
  unsigned short* a_cat2 = (unsigned short*)carve((size_t)B * 128 * HW1 * 2); // [up1_act | f1b]
  unsigned short* a_f2a  = (unsigned short*)carve((size_t)B * 128 * HW2 * 2);
  unsigned short* a_f2b  = (unsigned short*)carve((size_t)B * 128 * HW2 * 2);
  unsigned short* a_up1  = (unsigned short*)carve((size_t)B * 64 * HW * 2);   // final unet feature
  float* mean = (float*)carve(128 * 4);
  float* var  = (float*)carve(128 * 4);
  float* scal = (float*)carve(256);

  auto cdiv = [](int a, int b) { return (a + b - 1) / b; };

  // ---- weights -> bf16 [K][N] ----
  auto wc = [&](const float* w, unsigned short* dst, int Cin, int Cout, int tr) {
    int tot = Cin * 9 * Cout;
    wcvt_kernel<<<cdiv(tot, 256), 256, 0, stream>>>(w, dst, Cin, Cout, tr);
  };
  wc(c1a_w, wb_c1a, 132, 64, 0);
  wc(c1b_w, wb_c1b, 64, 64, 0);
  wc(c2a_w, wb_c2a, 64, 128, 0);
  wc(c2b_w, wb_c2b, 128, 128, 0);
  wc(up1_w, wb_up1, 128, 64, 1);
  wc(up2_w, wb_up2, 128, 64, 1);
  wc(off_w, wb_off, 64, 2, 0);

  // ---- warp + build U-Net input ----
  build_unet_input<<<cdiv(BHW, 256), 256, 0, stream>>>(lf, li, rf, ri, disp, xin, B, H, W);

  auto conv = [&](const unsigned short* in, const unsigned short* w, float* outp,
                  int Cin, int Hin, int Win, int Cout, int Hout, int Wout,
                  int stride, int tr) {
    int Mtot = B * Hout * Wout;
    dim3 grid(cdiv(Mtot, 64), cdiv(Cout, 64));
    conv_wmma_kernel<<<grid, 128, 0, stream>>>(in, w, outp, B, Cin, Hin, Win,
                                               Cout, Hout, Wout, stride, tr);
  };
  auto bnact = [&](float* rawp, const float* g, const float* bt, int Cc, int HWl,
                   unsigned short* dst, int dstC, int c_off,
                   unsigned short* dst2, int dst2C, int c_off2, int leaky) {
    bn_stats<<<Cc, 256, 0, stream>>>(rawp, mean, var, B, Cc, HWl);
    int tot = B * Cc * HWl;
    bn_norm_act<<<cdiv(tot, 256), 256, 0, stream>>>(rawp, mean, var, g, bt,
                                                    dst, dstC, c_off, dst2, dst2C,
                                                    c_off2, B, Cc, HWl, leaky);
  };

  // ---- U-Net ----
  conv(xin, wb_c1a, raw, 132, H, W, 64, H1, W1, 2, 0);
  bnact(raw, c1a_g, c1a_b, 64, HW1, a_f1a, 64, 0, nullptr, 0, 0, 0);

  conv(a_f1a, wb_c1b, raw, 64, H1, W1, 64, H1, W1, 1, 0);
  bnact(raw, c1b_g, c1b_b, 64, HW1, a_f1b, 64, 0, a_cat2, 128, 64, 0);  // f1b + concat copy

  conv(a_f1b, wb_c2a, raw, 64, H1, W1, 128, H2, W2, 2, 0);
  bnact(raw, c2a_g, c2a_b, 128, HW2, a_f2a, 128, 0, nullptr, 0, 0, 0);

  conv(a_f2a, wb_c2b, raw, 128, H2, W2, 128, H2, W2, 1, 0);
  bnact(raw, c2b_g, c2b_b, 128, HW2, a_f2b, 128, 0, nullptr, 0, 0, 0);

  conv(a_f2b, wb_up1, raw, 128, H2, W2, 64, H1, W1, 1, 1);              // convT
  bnact(raw, up1_g, up1_b, 64, HW1, a_cat2, 128, 0, nullptr, 0, 0, 1);  // lrelu -> concat[0:64]

  conv(a_cat2, wb_up2, raw, 128, H1, W1, 64, H, W, 1, 1);               // convT
  bnact(raw, up2_g, up2_b, 64, HW, a_up1, 64, 0, nullptr, 0, 0, 1);     // lrelu

  // ---- offset head ----
  conv(a_up1, wb_off, raw, 64, H, W, 2, H, W, 1, 0);
  off_bias_relu<<<cdiv(BHW, 256), 256, 0, stream>>>(raw, off_b, out_low, out_up, B, HW);

  // ---- max(lower), max(upper) ----
  zero_scal<<<1, 32, 0, stream>>>(scal);
  int rb = min(1024, cdiv(BHW, 256));
  reduce_max<<<rb, 256, 0, stream>>>(out_low, BHW, scal + 0);
  reduce_max<<<rb, 256, 0, stream>>>(out_up, BHW, scal + 1);

  // ---- cost volume + softmax residual ----
  cost_volume<<<cdiv(BHW, 256), 256, 0, stream>>>(lf, rf, disp, out_low, out_up,
                                                  scal, out_acv, out_mask,
                                                  B, C, H, W, R);
  softmax_residual<<<cdiv(BHW, 256), 256, 0, stream>>>(out_acv, scal, out_res, B, HW, R);
}
